// MultiHeadAttention_89361089561181
// MI455X (gfx1250) — matrix operand order repaired and checked by reading, compiled, not run on hardware
//
#include <hip/hip_runtime.h>
#include <hip/hip_bf16.h>

#define B_  4
#define S_  2048
#define D_  1024
#define H_  16
#define DK_ 64

typedef __bf16 bf16_t;
typedef __attribute__((ext_vector_type(16))) __bf16 v16bf;
typedef __attribute__((ext_vector_type(8)))  __bf16 v8bf;
typedef __attribute__((ext_vector_type(8)))  float  v8f;
typedef __attribute__((ext_vector_type(4)))  int    v4i;

typedef __attribute__((address_space(1))) v4i as1_v4i;
typedef __attribute__((address_space(3))) v4i as3_v4i;

#if __has_builtin(__builtin_amdgcn_global_load_async_to_lds_b128) && \
    __has_builtin(__builtin_amdgcn_s_wait_asynccnt)
#define MHA_USE_ASYNC 1
#else
#define MHA_USE_ASYNC 0
#endif

// ---- WMMA fragment loaders (CDNA5 ISA 7.12.2 layouts, wave32) ----

// A: 16x32 bf16 tile, row-major, lda elements.
// lanes 0-15: M=lane, K = {0..7, 16..23}; lanes 16-31: M=lane-16, K = {8..15, 24..31}
__device__ __forceinline__ v16bf load_a_bf16(const bf16_t* __restrict__ tile, int lda, int lane) {
    const int m = lane & 15, hf = lane >> 4;
    const v8bf lo = *reinterpret_cast<const v8bf*>(tile + m * lda + hf * 8);
    const v8bf hi = *reinterpret_cast<const v8bf*>(tile + m * lda + 16 + hf * 8);
    v16bf a;
#pragma unroll
    for (int i = 0; i < 8; ++i) { a[i] = lo[i]; a[8 + i] = hi[i]; }
    return a;
}

// Same as above but source is f32 (convert to bf16 in registers).
__device__ __forceinline__ v16bf load_a_f32(const float* __restrict__ tile, int lda, int lane) {
    const int m = lane & 15, hf = lane >> 4;
    const v8f lo = *reinterpret_cast<const v8f*>(tile + m * lda + hf * 8);
    const v8f hi = *reinterpret_cast<const v8f*>(tile + m * lda + 16 + hf * 8);
    v16bf a;
#pragma unroll
    for (int i = 0; i < 8; ++i) { a[i] = (bf16_t)lo[i]; a[8 + i] = (bf16_t)hi[i]; }
    return a;
}

// B: 32x16 bf16 tile B[k][n] where column n is contiguous in memory:
// element = tile[n*ldb + k].  lanes 0-15: N=lane, K = {0..7, 16..23}; lanes 16-31: N=lane-16, K = {8..15, 24..31}
__device__ __forceinline__ v16bf load_b_kmajor(const bf16_t* __restrict__ tile, int ldb, int lane) {
    const int n = lane & 15, kg = lane >> 4;
    const v8bf lo = *reinterpret_cast<const v8bf*>(tile + n * ldb + kg * 8);
    const v8bf hi = *reinterpret_cast<const v8bf*>(tile + n * ldb + 16 + kg * 8);
    v16bf b;
#pragma unroll
    for (int i = 0; i < 8; ++i) { b[i] = lo[i]; b[8 + i] = hi[i]; }
    return b;
}

#define WMMA_BF16(a, b, c) \
    __builtin_amdgcn_wmma_f32_16x16x32_bf16(false, (a), false, (b), (short)0, (c), false, false)

// 16-byte global->LDS stage (async on CDNA5 path, sync fallback)
__device__ __forceinline__ void stage16(const bf16_t* g, bf16_t* l) {
#if MHA_USE_ASYNC
    __builtin_amdgcn_global_load_async_to_lds_b128((as1_v4i*)g, (as3_v4i*)l, 0, 0);
#else
    *reinterpret_cast<v8bf*>(l) = *reinterpret_cast<const v8bf*>(g);
#endif
}

__device__ __forceinline__ void stage_wait() {
#if MHA_USE_ASYNC
    __builtin_amdgcn_s_wait_asynccnt(0);
#endif
}

// ---------------- f32 -> bf16 convert ----------------
__global__ void __launch_bounds__(256)
mha_cvt_bf16_kernel(const float* __restrict__ in, bf16_t* __restrict__ out, int n) {
    int i = blockIdx.x * 256 + threadIdx.x;
    if (i < n) out[i] = (bf16_t)in[i];
}

// ---------------- projection GEMM:  C = A @ W^T (+bias) ----------------
// Block tile 128x64 (8 waves x 16 rows, 64 shared cols), B tile double-buffered
// through LDS via async-to-LDS; each wave: 1 A fragment feeds 4 WMMAs per k-step.
// mode 0: bf16 (B,H,S,DK); mode 1: bf16 V^T (B,H,DK,S); mode 2: f32 out[t*D_+n].
#define BM_ 128
#define BN_ 64
#define KSTEP_ 32
#define BSTRIDE_ 40   // bf16 elems per LDS row (80 B, pad kills ds bank conflicts)

__global__ void __launch_bounds__(256)
mha_proj_kernel(const bf16_t* __restrict__ A, const bf16_t* __restrict__ W,
                const float* __restrict__ bias,
                bf16_t* __restrict__ out_bf, float* __restrict__ out_f,
                int M, int Kdim, int mode)
{
    __shared__ bf16_t Blds[2][BN_ * BSTRIDE_];

    const int lane = threadIdx.x & 31;
    const int wave = threadIdx.x >> 5;

    const int nBlocksN = D_ / BN_;                  // 16
    const int mblk = blockIdx.x / nBlocksN;
    const int nblk = blockIdx.x % nBlocksN;
    if (mblk * BM_ >= M) return;

    const bf16_t* Abase = A + ((size_t)mblk * BM_ + wave * 16) * Kdim;
    const bf16_t* Wbase = W + (size_t)nblk * BN_ * Kdim;

    // staging: thread t copies 16B chunk (t&3) of row (t>>2) of the 64x32 B tile
    const int srow = threadIdx.x >> 2;
    const int schunk = threadIdx.x & 3;
    const bf16_t* gsrc = Wbase + srow * Kdim + schunk * 8;
    bf16_t* ldst[2] = { &Blds[0][srow * BSTRIDE_ + schunk * 8],
                        &Blds[1][srow * BSTRIDE_ + schunk * 8] };

    v8f acc[4] = {};
    const int nIter = Kdim / KSTEP_;

    stage16(gsrc, ldst[0]);                          // prologue: k0 = 0 -> buf 0
    for (int it = 0; it < nIter; ++it) {
        stage_wait();
        __syncthreads();                             // buf[it&1] ready for all waves
        if (it + 1 < nIter)
            stage16(gsrc + (it + 1) * KSTEP_, ldst[(it + 1) & 1]);

        const int k0 = it * KSTEP_;
        if (k0 + 256 < Kdim)
            __builtin_prefetch((const void*)(Abase + k0 + 256), 0, 3);

        v16bf a = load_a_bf16(Abase + k0, Kdim, lane);
#pragma unroll
        for (int s4 = 0; s4 < 4; ++s4) {
            v16bf b = load_b_kmajor(&Blds[it & 1][s4 * 16 * BSTRIDE_], BSTRIDE_, lane);
            acc[s4] = WMMA_BF16(a, b, acc[s4]);
        }
        __syncthreads();                             // reads done before buf reuse
    }

    const int hf = lane >> 4, ncol = lane & 15;
#pragma unroll
    for (int s4 = 0; s4 < 4; ++s4) {
        const int n = nblk * BN_ + s4 * 16 + ncol;
        const float bv = bias ? bias[n] : 0.0f;
        const int h = n >> 6, dk = n & 63;
#pragma unroll
        for (int i = 0; i < 8; ++i) {
            const int t = mblk * BM_ + wave * 16 + hf * 8 + i;
            const float val = acc[s4][i] + bv;
            if (mode == 2) {
                out_f[(size_t)t * D_ + n] = val;
            } else {
                const int bb = t / S_, s = t % S_;
                if (mode == 0)
                    out_bf[(((size_t)bb * H_ + h) * S_ + s) * DK_ + dk] = (bf16_t)val;
                else
                    out_bf[(((size_t)bb * H_ + h) * DK_ + dk) * S_ + s] = (bf16_t)val;
            }
        }
    }
}

// ---------------- scores = (Q K^T) * 1/sqrt(DK), masked ----------------
__global__ void __launch_bounds__(256)
mha_scores_kernel(const bf16_t* __restrict__ Q, const bf16_t* __restrict__ Kmat,
                  const int* __restrict__ mask, float* __restrict__ wts)
{
    const int lane = threadIdx.x & 31;
    const int wave = threadIdx.x >> 5;
    int tid = blockIdx.x * 8 + wave;
    const int kt = tid & 127; tid >>= 7;
    const int qt = tid & 127; tid >>= 7;
    const int h  = tid & (H_ - 1);
    const int b  = tid >> 4;

    const bf16_t* Qb = Q    + (((size_t)b * H_ + h) * S_ + qt * 16) * DK_;
    const bf16_t* Kb = Kmat + (((size_t)b * H_ + h) * S_ + kt * 16) * DK_;

    v8f c = {};
#pragma unroll
    for (int k0 = 0; k0 < DK_; k0 += 32) {
        v16bf a = load_a_bf16(Qb + k0, DK_, lane);
        v16bf bfr = load_b_kmajor(Kb + k0, DK_, lane);
        c = WMMA_BF16(a, bfr, c);
    }

    const int hf = lane >> 4, ncol = lane & 15;
    const int kk = kt * 16 + ncol;
    const int mv = mask[b * S_ + kk];
    float* wrow = wts + (((size_t)b * H_ + h) * S_ + qt * 16 + hf * 8) * S_ + kk;
#pragma unroll
    for (int i = 0; i < 8; ++i) {
        float v = (mv == 0) ? -1.0e9f : c[i] * 0.125f;   // 1/sqrt(64)
        wrow[(size_t)i * S_] = v;
    }
}

// ---------------- row softmax over S_ elements, in place ----------------
__global__ void __launch_bounds__(256)
mha_softmax_kernel(float* __restrict__ wts)
{
    __shared__ float red[256];
    float* p = wts + (size_t)blockIdx.x * S_;
    const int t = threadIdx.x;

    float vals[8];
    float mx = -3.0e38f;
#pragma unroll
    for (int i = 0; i < 8; ++i) { vals[i] = p[t + i * 256]; mx = fmaxf(mx, vals[i]); }
    red[t] = mx; __syncthreads();
    for (int off = 128; off > 0; off >>= 1) {
        if (t < off) red[t] = fmaxf(red[t], red[t + off]);
        __syncthreads();
    }
    mx = red[0]; __syncthreads();

    float sum = 0.0f;
#pragma unroll
    for (int i = 0; i < 8; ++i) { vals[i] = __expf(vals[i] - mx); sum += vals[i]; }
    red[t] = sum; __syncthreads();
    for (int off = 128; off > 0; off >>= 1) {
        if (t < off) red[t] += red[t + off];
        __syncthreads();
    }
    const float inv = 1.0f / red[0];
#pragma unroll
    for (int i = 0; i < 8; ++i) p[t + i * 256] = vals[i] * inv;
}

// ---------------- attn = weights @ V  (V stored transposed) ----------------
// writes attn merged to (B, S, D) bf16 for the output projection
__global__ void __launch_bounds__(256)
mha_pv_kernel(const float* __restrict__ wts, const bf16_t* __restrict__ Vt,
              bf16_t* __restrict__ attn)
{
    const int lane = threadIdx.x & 31;
    const int wave = threadIdx.x >> 5;
    int tid = blockIdx.x * 8 + wave;
    const int dt = tid & 3;   tid >>= 2;
    const int qt = tid & 127; tid >>= 7;
    const int h  = tid & (H_ - 1);
    const int b  = tid >> 4;

    const float*  Abase = wts + (((size_t)b * H_ + h) * S_ + qt * 16) * S_;
    const bf16_t* Bbase = Vt  + (((size_t)b * H_ + h) * DK_ + dt * 16) * S_;

    v8f c = {};
    for (int k0 = 0; k0 < S_; k0 += 32) {
        if (k0 + 128 < S_) {
            __builtin_prefetch((const void*)(Abase + k0 + 128), 0, 3);
            __builtin_prefetch((const void*)(Bbase + k0 + 128), 0, 3);
        }
        v16bf a = load_a_f32(Abase + k0, S_, lane);
        v16bf bfr = load_b_kmajor(Bbase + k0, S_, lane);
        c = WMMA_BF16(a, bfr, c);
    }

    const int hf = lane >> 4, ncol = lane & 15;
    const int dg = h * DK_ + dt * 16 + ncol;
#pragma unroll
    for (int i = 0; i < 8; ++i) {
        const int s = qt * 16 + hf * 8 + i;
        attn[((size_t)b * S_ + s) * D_ + dg] = (bf16_t)c[i];
    }
}

extern "C" void kernel_launch(void* const* d_in, const int* in_sizes, int n_in,
                              void* d_out, int out_size, void* d_ws, size_t ws_size,
                              hipStream_t stream) {
    (void)in_sizes; (void)n_in; (void)out_size; (void)ws_size;

    const float* x    = (const float*)d_in[0];
    const int*   mask = (const int*)  d_in[1];
    const float* Wq   = (const float*)d_in[2];
    const float* bq   = (const float*)d_in[3];
    const float* Wk   = (const float*)d_in[4];
    const float* bk   = (const float*)d_in[5];
    const float* Wv   = (const float*)d_in[6];
    const float* bv   = (const float*)d_in[7];
    const float* Wo   = (const float*)d_in[8];
    const float* bo   = (const float*)d_in[9];

    float* out = (float*)d_out;                       // (B, S, D) f32
    float* wts = out + (size_t)B_ * S_ * D_;          // (B, H, S, S) f32

    // ---- workspace carve-up (bf16 intermediates) ----
    char* ws = (char*)d_ws;
    const size_t SZ_X  = (size_t)B_ * S_ * D_ * sizeof(bf16_t);       // 16 MiB
    const size_t SZ_W  = (size_t)D_ * D_ * sizeof(bf16_t);            //  2 MiB
    const size_t SZ_H  = (size_t)B_ * H_ * S_ * DK_ * sizeof(bf16_t); // 16 MiB
    bf16_t* xb   = (bf16_t*)(ws);
    bf16_t* Wqb  = (bf16_t*)(ws + SZ_X);
    bf16_t* Wkb  = (bf16_t*)(ws + SZ_X + SZ_W);
    bf16_t* Wvb  = (bf16_t*)(ws + SZ_X + 2 * SZ_W);
    bf16_t* Wob  = (bf16_t*)(ws + SZ_X + 3 * SZ_W);
    bf16_t* Qb   = (bf16_t*)(ws + SZ_X + 4 * SZ_W);
    bf16_t* Kb   = (bf16_t*)(ws + SZ_X + 4 * SZ_W + SZ_H);
    bf16_t* Vtb  = (bf16_t*)(ws + SZ_X + 4 * SZ_W + 2 * SZ_H);
    bf16_t* attn = (bf16_t*)(ws + SZ_X + 4 * SZ_W + 3 * SZ_H);

    const int M = B_ * S_;                 // 8192 tokens
    const int nX = M * D_;
    const int nW = D_ * D_;

    // 1) converts
    mha_cvt_bf16_kernel<<<(nX + 255) / 256, 256, 0, stream>>>(x,  xb,  nX);
    mha_cvt_bf16_kernel<<<(nW + 255) / 256, 256, 0, stream>>>(Wq, Wqb, nW);
    mha_cvt_bf16_kernel<<<(nW + 255) / 256, 256, 0, stream>>>(Wk, Wkb, nW);
    mha_cvt_bf16_kernel<<<(nW + 255) / 256, 256, 0, stream>>>(Wv, Wvb, nW);
    mha_cvt_bf16_kernel<<<(nW + 255) / 256, 256, 0, stream>>>(Wo, Wob, nW);

    // 2) Q, K, V^T projections: (M/128)*(D/64) = 1024 blocks of 8 waves
    const int projBlocks = (M / BM_) * (D_ / BN_);
    mha_proj_kernel<<<projBlocks, 256, 0, stream>>>(xb, Wqb, bq, Qb,  nullptr, M, D_, 0);
    mha_proj_kernel<<<projBlocks, 256, 0, stream>>>(xb, Wkb, bk, Kb,  nullptr, M, D_, 0);
    mha_proj_kernel<<<projBlocks, 256, 0, stream>>>(xb, Wvb, bv, Vtb, nullptr, M, D_, 1);

    // 3) masked scaled scores -> weights buffer
    const int scoreBlocks = (B_ * H_ * (S_ / 16) * (S_ / 16)) / 8;
    mha_scores_kernel<<<scoreBlocks, 256, 0, stream>>>(Qb, Kb, mask, wts);

    // 4) softmax in place: one block per row
    mha_softmax_kernel<<<B_ * H_ * S_, 256, 0, stream>>>(wts);

    // 5) attn = weights @ V
    const int pvBlocks = (B_ * H_ * (S_ / 16) * (DK_ / 16)) / 8;
    mha_pv_kernel<<<pvBlocks, 256, 0, stream>>>(wts, Vtb, attn);

    // 6) out = attn @ Wo^T + bo
    mha_proj_kernel<<<projBlocks, 256, 0, stream>>>(attn, Wob, bo, nullptr, out, M, D_, 2);
}
